// metapath_preference_attention_35158602285298
// MI455X (gfx1250) — matrix operand order repaired and checked by reading, not compiled, not run on hardware
//
#include <hip/hip_runtime.h>
#include <cstdint>
#include <cstddef>

typedef __attribute__((ext_vector_type(16))) __bf16 v16bf;
typedef __attribute__((ext_vector_type(8)))  float  v8f;

#define M_      4
#define TABLE_  300000
#define N_      200000
#define D_      128
#define P_      64

#define OSTRIDE 132   // f32 elements per orig row in LDS (128 + 4 pad)
#define WSTRIDE 132   // bf16 elements per W row in LDS (128 + 4 pad)

// branch-free tanh: 1 - 2/(exp(2x)+1); exact at both saturation limits
__device__ __forceinline__ float fast_tanh(float x) {
    const float ex = __expf(2.0f * x);
    return 1.0f - 2.0f * __builtin_amdgcn_rcpf(ex + 1.0f);
}

__global__ __launch_bounds__(128) void metapath_attn_kernel(
    const float* __restrict__ emb,        // [M, TABLE, D] f32
    const long long* __restrict__ idxmap, // [M, N] i64
    const float* __restrict__ W,          // [P, D] f32
    const float* __restrict__ bias,       // [P] f32
    const float* __restrict__ pref,       // [M, P] f32
    float* __restrict__ out)              // [N, D] f32
{
    __shared__ __attribute__((aligned(16))) __bf16 ldsW[P_ * WSTRIDE];
    __shared__ __attribute__((aligned(16))) float  ldsOrig[4 * 16 * OSTRIDE];
    __shared__ float ldsB[P_];
    __shared__ float ldsPref[M_ * P_];
    __shared__ float ldsN2[M_];

    const int tid  = threadIdx.x;
    const int wave = tid >> 5;
    const int lane = tid & 31;

    // ---- gather 16 rows (4 n x 4 m), row = nl*4 + m, straight into LDS ----
    // CDNA5 async copy: global -> LDS without VGPR staging, tracked by ASYNCcnt.
    const int n0 = (blockIdx.x * 4 + wave) * 4;
    float* myOrig = ldsOrig + wave * 16 * OSTRIDE;
    #pragma unroll
    for (int it = 0; it < 16; ++it) {
        const int m  = it & 3;
        const int nl = it >> 2;
        const long long idx = idxmap[(size_t)m * N_ + (size_t)(n0 + nl)];
        const float* src = emb + ((size_t)m * TABLE_ + (size_t)idx) * D_ + lane * 4;
        // low 32 bits of the flat shared address == LDS byte address (aperture rule)
        const unsigned ldsoff = (unsigned)(size_t)(const void*)(myOrig + it * OSTRIDE + lane * 4);
        asm volatile("global_load_async_to_lds_b128 %0, %1, off"
                     :: "v"(ldsoff), "v"(src)
                     : "memory");
    }

    // ---- preload W (f32->bf16), bias, preference, preference norms ----
    for (int i = tid; i < P_ * D_; i += 128) {
        const int r = i >> 7, c = i & 127;
        ldsW[r * WSTRIDE + c] = (__bf16)W[i];
    }
    if (tid < P_) ldsB[tid] = bias[tid];
    for (int i = tid; i < M_ * P_; i += 128) ldsPref[i] = pref[i];
    if (tid < M_) {
        float s = 0.f;
        for (int p = 0; p < P_; ++p) { const float v = pref[tid * P_ + p]; s += v * v; }
        ldsN2[tid] = sqrtf(s);
    }

    asm volatile("s_wait_asynccnt 0" ::: "memory");
    __syncthreads();

    // ---- build bf16 A fragments (16x32 layout, K=128 -> 4 fragments) ----
    const int rowM = lane & 15;
    const int hi   = lane >> 4;
    const float* rowPtr = myOrig + rowM * OSTRIDE;
    v16bf afrag[4];
    #pragma unroll
    for (int ks = 0; ks < 4; ++ks) {
        #pragma unroll
        for (int j = 0; j < 8; ++j) {
            const int K = 32 * ks + ((j < 4) ? 2 * j : 16 + 2 * (j - 4)) + 8 * hi;
            afrag[ks][2 * j]     = (__bf16)rowPtr[K];
            afrag[ks][2 * j + 1] = (__bf16)rowPtr[K + 1];
        }
    }

    // ---- GEMM (4 P-tiles x 4 k-steps of v_wmma_f32_16x16x32_bf16) ----
    float dotA[8], nsqA[8];
    #pragma unroll
    for (int r = 0; r < 8; ++r) { dotA[r] = 0.f; nsqA[r] = 0.f; }
    const int col = lane & 15;

    #pragma unroll
    for (int t = 0; t < 4; ++t) {
        const int p = 16 * t + col;
        const unsigned* wrow = (const unsigned*)(ldsW + p * WSTRIDE); // packed bf16 pairs
        v8f acc = {0.f, 0.f, 0.f, 0.f, 0.f, 0.f, 0.f, 0.f};
        #pragma unroll
        for (int ks = 0; ks < 4; ++ks) {
            union { unsigned u[8]; v16bf v; } bf;
            #pragma unroll
            for (int j = 0; j < 8; ++j) bf.u[j] = wrow[16 * ks + 4 * hi + j + ((j < 4) ? 0 : 4)];
            acc = __builtin_amdgcn_wmma_f32_16x16x32_bf16(
                false, afrag[ks], false, bf.v, (short)0, acc, false, false);
        }
        // tanh(acc + b[p]); accumulate dot with pref and squared norm per row
        const float bp = ldsB[p];
        #pragma unroll
        for (int r = 0; r < 8; ++r) {
            const float tr = fast_tanh(acc[r] + bp);
            const int m = r & 3;                 // global row = 8*hi + r; m = row & 3
            dotA[r] += tr * ldsPref[m * P_ + p];
            nsqA[r] += tr * tr;
        }
    }

    // ---- reduce over the 16 columns (P dim) within each half-wave ----
    #pragma unroll
    for (int r = 0; r < 8; ++r) {
        float d = dotA[r], q = nsqA[r];
        #pragma unroll
        for (int off = 1; off < 16; off <<= 1) {
            d += __shfl_xor(d, off, 16);
            q += __shfl_xor(q, off, 16);
        }
        dotA[r] = d; nsqA[r] = q;
    }

    // ---- cosine sim, softmax over m, weighted f32 sum -> out ----
    float n2v[4];
    #pragma unroll
    for (int m = 0; m < 4; ++m) n2v[m] = ldsN2[m];

    #pragma unroll
    for (int nl2 = 0; nl2 < 2; ++nl2) {   // two n's per half-wave
        float sim[4];
        #pragma unroll
        for (int m = 0; m < 4; ++m) {
            const int lr = nl2 * 4 + m;
            sim[m] = dotA[lr] / fmaxf(sqrtf(nsqA[lr]) * n2v[m], 1e-8f);
        }
        const float mx = fmaxf(fmaxf(sim[0], sim[1]), fmaxf(sim[2], sim[3]));
        const float e0 = __expf(sim[0] - mx), e1 = __expf(sim[1] - mx);
        const float e2 = __expf(sim[2] - mx), e3 = __expf(sim[3] - mx);
        const float inv = 1.0f / (e0 + e1 + e2 + e3);
        const float w0 = e0 * inv, w1 = e1 * inv, w2 = e2 * inv, w3 = e3 * inv;

        const int nloc = 2 * hi + nl2;
        const float* rows = myOrig + nloc * 4 * OSTRIDE;
        float* dst = out + (size_t)(n0 + nloc) * D_;
        #pragma unroll
        for (int s = 0; s < 2; ++s) {
            const int d0 = (lane & 15) * 4 + s * 64;
            const float4 v0 = *(const float4*)(rows + 0 * OSTRIDE + d0);
            const float4 v1 = *(const float4*)(rows + 1 * OSTRIDE + d0);
            const float4 v2 = *(const float4*)(rows + 2 * OSTRIDE + d0);
            const float4 v3 = *(const float4*)(rows + 3 * OSTRIDE + d0);
            float4 o;
            o.x = w0 * v0.x + w1 * v1.x + w2 * v2.x + w3 * v3.x;
            o.y = w0 * v0.y + w1 * v1.y + w2 * v2.y + w3 * v3.y;
            o.z = w0 * v0.z + w1 * v1.z + w2 * v2.z + w3 * v3.z;
            o.w = w0 * v0.w + w1 * v1.w + w2 * v2.w + w3 * v3.w;
            *(float4*)(dst + d0) = o;
        }
    }
}

extern "C" void kernel_launch(void* const* d_in, const int* in_sizes, int n_in,
                              void* d_out, int out_size, void* d_ws, size_t ws_size,
                              hipStream_t stream) {
    (void)in_sizes; (void)n_in; (void)out_size; (void)d_ws; (void)ws_size;
    const float*     emb    = (const float*)d_in[0];
    const long long* idxmap = (const long long*)d_in[1];
    const float*     W      = (const float*)d_in[2];
    const float*     bias   = (const float*)d_in[3];
    const float*     pref   = (const float*)d_in[4];
    float*           out    = (float*)d_out;

    // 128 threads = 4 waves; each wave handles 4 n -> 16 n per block
    const int blocks = N_ / 16;  // 200000 / 16 = 12500 exactly
    metapath_attn_kernel<<<blocks, 128, 0, stream>>>(emb, idxmap, W, bias, pref, out);
}
